// KronyMLP_49984829390894
// MI455X (gfx1250) — matrix operand order repaired and checked by reading, not compiled, not run on hardware
//
#include <hip/hip_runtime.h>
#include <cmath>

typedef __attribute__((ext_vector_type(16))) _Float16 v16h;
typedef __attribute__((ext_vector_type(8)))  _Float16 half8;
typedef __attribute__((ext_vector_type(8)))  float    v8f;

#define D_MODEL 384
#define D_FF    1536
#define KDIM    32      // inner dim of both small GEMMs (cols of c_fc_1 / rows of c_proj_1)
#define RDIM    12      // Kronecker rank-1 factor length
#define M_TILE  128     // rows per workgroup
#define CHUNK   128     // h columns per LDS chunk
#define NCHUNK  (D_FF / CHUNK)   // 12

__device__ __forceinline__ float gelu_erf(float v) {
    // exact erf GELU (reference uses approximate=False)
    return 0.5f * v * (1.0f + erff(v * 0.7071067811865475f));
}

// Convert 16 consecutive floats -> 16 f16 in LDS (two 16B stores)
__device__ __forceinline__ void cvt_store16(const float4* __restrict__ p, _Float16* dst) {
    float4 f0 = p[0], f1 = p[1], f2 = p[2], f3 = p[3];
    half8 h0, h1;
    h0[0]=(_Float16)f0.x; h0[1]=(_Float16)f0.y; h0[2]=(_Float16)f0.z; h0[3]=(_Float16)f0.w;
    h0[4]=(_Float16)f1.x; h0[5]=(_Float16)f1.y; h0[6]=(_Float16)f1.z; h0[7]=(_Float16)f1.w;
    h1[0]=(_Float16)f2.x; h1[1]=(_Float16)f2.y; h1[2]=(_Float16)f2.z; h1[3]=(_Float16)f2.w;
    h1[4]=(_Float16)f3.x; h1[5]=(_Float16)f3.y; h1[6]=(_Float16)f3.z; h1[7]=(_Float16)f3.w;
    *(half8*)dst = h0;
    *(half8*)(dst + 8) = h1;
}

__global__ __launch_bounds__(256)
void krony_mlp_fused(const float* __restrict__ x,
                     const float* __restrict__ cfc1,   // [1536,32]
                     const float* __restrict__ cfc2,   // [1,12]
                     const float* __restrict__ cpr1,   // [32,1536]
                     const float* __restrict__ cpr2,   // [12,1]
                     float* __restrict__ out) {
    extern __shared__ _Float16 smem[];
    _Float16* yLds   = smem;                               // [128][32]   8 KB
    _Float16* hLds   = yLds + M_TILE * KDIM;               // [128][128] 32 KB
    _Float16* wfcLds = hLds + M_TILE * CHUNK;              // [128][32]   8 KB
    _Float16* wprLds = wfcLds + CHUNK * KDIM;              // [32][128]   8 KB

    const int tid   = threadIdx.x;
    const int wave  = tid >> 5;     // 0..7 (wave32)
    const int lane  = tid & 31;
    const int laneM = lane & 15;
    const int laneH = lane >> 4;
    const long m0   = (long)blockIdx.x * M_TILE;

    // tiny rank-1 vectors in registers
    float c2[RDIM], p2[RDIM];
    #pragma unroll
    for (int j = 0; j < RDIM; ++j) { c2[j] = cfc2[j]; p2[j] = cpr2[j]; }

    // ---- Phase 0: y[m,k] = sum_{s<12} x[m, 12k+s] * cfc2[s]  (fp32 -> f16 LDS)
    {
        const int m    = tid >> 1;          // 0..127
        const int half = tid & 1;           // k 0..15 / 16..31
        const float* xr = x + (m0 + m) * D_MODEL + half * (16 * RDIM);
        #pragma unroll
        for (int i = 0; i < 16; ++i) {
            const float4* p = (const float4*)(xr + i * RDIM);   // 48B, 16B-aligned
            float4 f0 = p[0], f1 = p[1], f2 = p[2];
            float acc = f0.x*c2[0] + f0.y*c2[1] + f0.z*c2[2]  + f0.w*c2[3]
                      + f1.x*c2[4] + f1.y*c2[5] + f1.z*c2[6]  + f1.w*c2[7]
                      + f2.x*c2[8] + f2.y*c2[9] + f2.z*c2[10] + f2.w*c2[11];
            yLds[m * KDIM + half * 16 + i] = (_Float16)acc;
        }
    }

    v8f g0 = {}, g1 = {};   // persistent accumulators g[m, 0..31] for my 16 rows

    for (int cc = 0; cc < NCHUNK; ++cc) {
        const int nbase = cc * CHUNK;

        // ---- Stage this chunk's weight slices into LDS as f16 (coalesced, once) ----
        {
            // c_fc_1 rows [nbase, nbase+128): 4096 floats, 16 per thread
            const int row = tid >> 1, half = tid & 1;
            cvt_store16((const float4*)(cfc1 + (size_t)(nbase + row) * KDIM + half * 16),
                        wfcLds + row * KDIM + half * 16);
            // c_proj_1[:, nbase..nbase+128): 4096 floats, 16 per thread
            const int n = tid >> 3, seg = tid & 7;
            cvt_store16((const float4*)(cpr1 + (size_t)n * D_FF + nbase + seg * 16),
                        wprLds + n * CHUNK + seg * 16);
            // warm caches for next chunk's weight slices (global_prefetch_b8)
            if (cc + 1 < NCHUNK) {
                const int nb2 = nbase + CHUNK;
                __builtin_prefetch(cfc1 + (size_t)(nb2 + row) * KDIM + half * 16, 0, 3);
                __builtin_prefetch(cpr1 + (size_t)n * D_FF + nb2 + seg * 16, 0, 3);
            }
        }
        __syncthreads();

        // ---- Phase A: h[:, nbase+16w .. +15] = gelu(y @ cfc1^T), K=32 (1 WMMA/tile)
        {
            // B tile (32x16) from LDS: B[k,n] = cfc1[nbase+16w+n, k]; elems 0..7 -> k=8h+0..7, elems 8..15 -> k=16+8h+0..7
            const int brow = wave * 16 + laneM;
            half8 blo = *(const half8*)(wfcLds + brow * KDIM + laneH * 8);
            half8 bhi = *(const half8*)(wfcLds + brow * KDIM + 16 + laneH * 8);
            v16h b;
            #pragma unroll
            for (int e = 0; e < 8; ++e) { b[e] = blo[e]; b[8 + e] = bhi[e]; }

            #pragma unroll
            for (int mt = 0; mt < 8; ++mt) {
                // A tile (16x32): elems 0..7 -> k=8h+0..7, elems 8..15 -> k=16+8h+0..7
                const int m = mt * 16 + laneM;
                half8 lo = *(const half8*)(yLds + m * KDIM + laneH * 8);
                half8 hi = *(const half8*)(yLds + m * KDIM + 16 + laneH * 8);
                v16h a;
                #pragma unroll
                for (int e = 0; e < 8; ++e) { a[e] = lo[e]; a[8 + e] = hi[e]; }
                v8f c = {};
                c = __builtin_amdgcn_wmma_f32_16x16x32_f16(false, a, false, b,
                                                           (short)0, c, false, false);
                // GELU on f32 accumulators, store f16 h-chunk to LDS
                const int col = wave * 16 + laneM;
                #pragma unroll
                for (int r = 0; r < 8; ++r) {
                    const int row = mt * 16 + r + 8 * laneH;   // C/D layout: M = r + 8*half
                    hLds[row * CHUNK + col] = (_Float16)gelu_erf(c[r]);
                }
            }
        }
        __syncthreads();

        // ---- Phase B: g += h_chunk[16w..16w+15, :] @ cpr1[:, chunk]^T  (K = CHUNK)
        {
            const int m = wave * 16 + laneM;
            #pragma unroll
            for (int s = 0; s < CHUNK / 32; ++s) {
                half8 lo = *(const half8*)(hLds + m * CHUNK + s * 32 + laneH * 8);
                half8 hi = *(const half8*)(hLds + m * CHUNK + s * 32 + 16 + laneH * 8);
                v16h a;
                #pragma unroll
                for (int e = 0; e < 8; ++e) { a[e] = lo[e]; a[8 + e] = hi[e]; }

                // B tiles (32x16) from LDS: B[k,n] = cpr1[n, nbase + k_local]
                const int kl = s * 32 + laneH * 8;
                half8 c0l = *(const half8*)(wprLds + laneM * CHUNK + kl);
                half8 c0h = *(const half8*)(wprLds + laneM * CHUNK + kl + 16);
                half8 c1l = *(const half8*)(wprLds + (16 + laneM) * CHUNK + kl);
                half8 c1h = *(const half8*)(wprLds + (16 + laneM) * CHUNK + kl + 16);
                v16h b0, b1;
                #pragma unroll
                for (int e = 0; e < 8; ++e) {
                    b0[e] = c0l[e]; b0[8 + e] = c0h[e];
                    b1[e] = c1l[e]; b1[8 + e] = c1h[e];
                }
                g0 = __builtin_amdgcn_wmma_f32_16x16x32_f16(false, a, false, b0,
                                                            (short)0, g0, false, false);
                g1 = __builtin_amdgcn_wmma_f32_16x16x32_f16(false, a, false, b1,
                                                            (short)0, g1, false, false);
            }
        }
        __syncthreads();   // protect hLds/weight slices before next chunk overwrites them
    }

    // ---- Final: out[m, 12k + r] = g[m,k] * cpr2[r]   (rank-1 Kron expansion on store)
    {
        #pragma unroll
        for (int r = 0; r < 8; ++r) {
            const long row = m0 + wave * 16 + r + 8 * laneH;   // C/D layout rows
            float* orow = out + row * D_MODEL;
            const float v0 = g0[r];            // k = laneM
            const float v1 = g1[r];            // k = 16 + laneM
            #pragma unroll
            for (int q = 0; q < 3; ++q) {
                float4 s0, s1;
                s0.x = v0 * p2[4*q+0]; s0.y = v0 * p2[4*q+1];
                s0.z = v0 * p2[4*q+2]; s0.w = v0 * p2[4*q+3];
                s1.x = v1 * p2[4*q+0]; s1.y = v1 * p2[4*q+1];
                s1.z = v1 * p2[4*q+2]; s1.w = v1 * p2[4*q+3];
                ((float4*)(orow + laneM        * RDIM))[q] = s0;   // 48B, 16B-aligned
                ((float4*)(orow + (16 + laneM) * RDIM))[q] = s1;
            }
        }
    }
}

extern "C" void kernel_launch(void* const* d_in, const int* in_sizes, int n_in,
                              void* d_out, int out_size, void* d_ws, size_t ws_size,
                              hipStream_t stream) {
    const float* x    = (const float*)d_in[0];
    const float* cfc1 = (const float*)d_in[1];
    const float* cfc2 = (const float*)d_in[2];
    const float* cpr1 = (const float*)d_in[3];
    const float* cpr2 = (const float*)d_in[4];
    float* out = (float*)d_out;

    const int rows = in_sizes[0] / D_MODEL;     // 16*2048 = 32768
    const int grid = rows / M_TILE;             // 256 workgroups
    const size_t shmem = (size_t)(M_TILE * KDIM + M_TILE * CHUNK +
                                  CHUNK * KDIM + KDIM * CHUNK) * sizeof(_Float16); // 56 KB

    krony_mlp_fused<<<grid, 256, shmem, stream>>>(x, cfc1, cfc2, cpr1, cpr2, out);
}
